// Mamba2HSIClassifier_41180146434143
// MI455X (gfx1250) — compile-verified, matrix operand order checked by reading, not run on hardware
//
#include <hip/hip_runtime.h>
#include <hip/hip_bf16.h>
#include <math.h>

// ---------------- model constants ----------------
#define HH 145
#define WW 145
#define BANDS 200
#define LSEQ (HH * WW)          // 21025
#define D_MODEL 128
#define NUM_CLASSES 17
#define D_INNER 256
#define NHEADS 4
#define HEADDIM 64
#define D_STATE 128
#define D_CONV 4
#define D_IN_PROJ (2 * D_INNER + 2 * D_STATE + NHEADS)   // 772
#define CONV_DIM (D_INNER + 2 * D_STATE)                 // 512
#define CH 147                  // conv spatial size (145 + 2*2 - 3 + 1)

typedef __attribute__((ext_vector_type(16))) _Float16 v16h;
typedef __attribute__((ext_vector_type(8)))  float    v8f;

__device__ __forceinline__ float gelu_f(float x) {
    return 0.5f * x * (1.0f + erff(x * 0.70710678118654752440f));
}
__device__ __forceinline__ float silu_f(float x) {
    return x / (1.0f + expf(-x));
}
__device__ __forceinline__ float softplus_f(float x) {
    return (x > 20.0f) ? x : log1pf(expf(x));
}

// ---------------- generic LayerNorm over rows (C <= 256), optional GELU ----------------
__global__ void __launch_bounds__(128)
ln_rows(const float* __restrict__ in, const float* __restrict__ g,
        const float* __restrict__ b, float* __restrict__ out, int C, int act)
{
    __shared__ float s1[128], s2[128];
    const int tid = threadIdx.x;
    const int base = (int)blockIdx.x * C;
    float xv[2];
    float sum = 0.f, sq = 0.f;
    int cnt = 0;
    for (int j = tid; j < C; j += 128) {
        float v = in[base + j];
        xv[cnt++] = v;
        sum += v; sq += v * v;
    }
    s1[tid] = sum; s2[tid] = sq;
    __syncthreads();
    for (int s = 64; s > 0; s >>= 1) {
        if (tid < s) { s1[tid] += s1[tid + s]; s2[tid] += s2[tid + s]; }
        __syncthreads();
    }
    const float m   = s1[0] / (float)C;
    const float var = s2[0] / (float)C - m * m;
    const float inv = rsqrtf(var + 1e-5f);
    cnt = 0;
    for (int j = tid; j < C; j += 128) {
        float v = (xv[cnt++] - m) * inv * g[j] + b[j];
        if (act == 1) v = gelu_f(v);
        out[base + j] = v;
    }
}

// ---------------- WMMA f16 GEMM, K compile-time: C[M,N](ldc) = act(A[M,K] @ B[K,N] + bias)
// 4 waves / block; wave w owns the 16x16 tile (tm = (by*4+w)*16, tn = bx*16).
// Branchless boundaries: OOB rows/cols load CLAMPED data (their D entries are never
// stored; A row m only feeds D row m, B col n only feeds D col n). Only K-OOB must be
// zeroed, and that is done once in the shared B staging; fully-OOB A runs (K=200 tail,
// runs are 8-aligned) load clamped garbage that multiplies zeroed B rows -> 0.
template <int K, int ACT, bool HAS_BIAS>
__global__ void __launch_bounds__(128)
gemm_wmma_t(const float* __restrict__ A, const float* __restrict__ Bw,
            const float* __restrict__ bias, float* __restrict__ C,
            int M, int N, int ldc)
{
    static_assert((K & 7) == 0, "K must be a multiple of 8");
    __shared__ alignas(16) float sB[32 * 16];
    const int lane  = threadIdx.x & 31;
    const int wave  = threadIdx.x >> 5;
    const int tm    = ((int)blockIdx.y * 4 + wave) * 16;
    const int tn    = (int)blockIdx.x * 16;
    const int row   = lane & 15;
    const int khalf = lane >> 4;
    const int am    = tm + row;
    const int amc   = (am < M) ? am : (M - 1);       // row clamp (OOB rows unstored)
    const float* __restrict__ Arow = A + amc * K;

    const int skk = threadIdx.x >> 2;                // 0..31 : k within slice
    const int sq  = (threadIdx.x & 3) * 4;           // 0,4,8,12 : n offset
    const bool nfull = (tn + 15) < N;                // uniform per block

    v8f acc = {};
    constexpr int KSTEPS = (K + 31) / 32;
    #pragma unroll
    for (int ks = 0; ks < KSTEPS; ++ks) {
        const int k0 = ks * 32;
        // ---- stage B slice (k0..k0+31) x 16 cols into LDS ----
        {
            const int  k   = k0 + skk;
            const bool kok = (k < K);                // folds except K=200 tail step
            const int  kcl = kok ? k : (K - 1);
            const int  n   = tn + sq;
            float4 bv;
            if (nfull) {
                bv = *(const float4*)(Bw + kcl * N + n);
            } else {                                 // partial-N tile (in_proj edge only)
                const float* Br = Bw + kcl * N;
                const int n0 = (n + 0 < N) ? n + 0 : N - 1;
                const int n1 = (n + 1 < N) ? n + 1 : N - 1;
                const int n2 = (n + 2 < N) ? n + 2 : N - 1;
                const int n3 = (n + 3 < N) ? n + 3 : N - 1;
                bv = make_float4(Br[n0], Br[n1], Br[n2], Br[n3]);
            }
            if (!kok) bv = make_float4(0.f, 0.f, 0.f, 0.f);   // only K-OOB zeroed
            *(float4*)(sB + skk * 16 + sq) = bv;
        }
        __syncthreads();

        if (ks + 1 < KSTEPS)
            __builtin_prefetch(Arow + k0 + 32, 0, 1);   // global_prefetch_b8

        // ---- fragments: A = two runs of 8 contiguous floats per lane (run-clamped) ----
        v16h af, bf;
        #pragma unroll
        for (int half = 0; half < 2; ++half) {
            const int kb  = k0 + 16 * half + 8 * khalf;
            const int kbc = (kb + 7 < K) ? kb : (K - 8);   // folds for interior steps
            const float4 a0 = *(const float4*)(Arow + kbc);
            const float4 a1 = *(const float4*)(Arow + kbc + 4);
            af[half * 8 + 0] = (_Float16)a0.x;
            af[half * 8 + 1] = (_Float16)a0.y;
            af[half * 8 + 2] = (_Float16)a0.z;
            af[half * 8 + 3] = (_Float16)a0.w;
            af[half * 8 + 4] = (_Float16)a1.x;
            af[half * 8 + 5] = (_Float16)a1.y;
            af[half * 8 + 6] = (_Float16)a1.z;
            af[half * 8 + 7] = (_Float16)a1.w;
            const int kbl = 16 * half + 8 * khalf;         // local slice row
            #pragma unroll
            for (int i = 0; i < 8; ++i)
                bf[half * 8 + i] = (_Float16)sB[(kbl + i) * 16 + row];
        }
        acc = __builtin_amdgcn_wmma_f32_16x16x32_f16(
                false, af, false, bf, (short)0, acc, false, false);
        __syncthreads();
    }

    // ---- epilogue: D layout lane n = lane&15, VGPR r -> m = tm + r + 8*khalf ----
    const int n = tn + row;
    const float bv = HAS_BIAS ? bias[(n < N) ? n : (N - 1)] : 0.0f;
    const int  m0 = tm + 8 * khalf;
    float* __restrict__ Crow = C + m0 * ldc + n;
    const bool mfull = (tm + 15) < M;                // uniform per wave
    if (mfull && n < N) {
        #pragma unroll
        for (int r = 0; r < 8; ++r) {
            float v = acc[r] + bv;
            if (ACT == 1) v = gelu_f(v);
            Crow[r * ldc] = v;
        }
    } else if (n < N) {
        #pragma unroll
        for (int r = 0; r < 8; ++r) {
            if (m0 + r < M) {
                float v = acc[r] + bv;
                if (ACT == 1) v = gelu_f(v);
                Crow[r * ldc] = v;
            }
        }
    }
}

// ---------------- conv1: (HWC in) 200ch -> 32ch, 3x3, pad 2 -> (32,147,147) + GELU ----------------
__global__ void __launch_bounds__(256)
conv1_gelu(const float* __restrict__ xn, const float* __restrict__ w,
           const float* __restrict__ b, float* __restrict__ out)
{
    const int HW = CH * CH;
    int idx = blockIdx.x * blockDim.x + threadIdx.x;
    if (idx >= 32 * HW) return;
    int o = idx / HW, rem = idx - o * HW;
    int y = rem / CH, x = rem - y * CH;
    float acc = b[o];
    for (int ky = 0; ky < 3; ++ky) {
        int yy = y - 2 + ky; if (yy < 0 || yy >= HH) continue;
        for (int kx = 0; kx < 3; ++kx) {
            int xx = x - 2 + kx; if (xx < 0 || xx >= WW) continue;
            const float* px = xn + (yy * WW + xx) * BANDS;
            const float* pw = w + o * BANDS * 9 + ky * 3 + kx;
            for (int c = 0; c < BANDS; ++c) acc += px[c] * pw[c * 9];
        }
    }
    out[idx] = gelu_f(acc);
}

// ---------------- conv2: 32 -> 64, 3x3, pad 1, CHW -> (64,147,147) + GELU ----------------
__global__ void __launch_bounds__(256)
conv2_gelu(const float* __restrict__ in, const float* __restrict__ w,
           const float* __restrict__ b, float* __restrict__ out)
{
    const int HW = CH * CH;
    int idx = blockIdx.x * blockDim.x + threadIdx.x;
    if (idx >= 64 * HW) return;
    int o = idx / HW, rem = idx - o * HW;
    int y = rem / CH, x = rem - y * CH;
    float acc = b[o];
    for (int c = 0; c < 32; ++c) {
        const float* pw = w + o * 32 * 9 + c * 9;
        const float* pc = in + c * HW;
        for (int ky = 0; ky < 3; ++ky) {
            int yy = y - 1 + ky; if (yy < 0 || yy >= CH) continue;
            for (int kx = 0; kx < 3; ++kx) {
                int xx = x - 1 + kx; if (xx < 0 || xx >= CH) continue;
                acc += pc[yy * CH + xx] * pw[ky * 3 + kx];
            }
        }
    }
    out[idx] = gelu_f(acc);
}

// ---------------- adaptive pool 147->145 (both dims) into comb cols [128..192) ----------------
__global__ void __launch_bounds__(256)
pool_to_comb(const float* __restrict__ c2, float* __restrict__ comb)
{
    int idx = blockIdx.x * blockDim.x + threadIdx.x;
    if (idx >= 64 * HH * WW) return;
    int c = idx / (HH * WW), rem = idx - c * HH * WW;
    int i = rem / WW, j = rem - i * WW;
    int hs = (i * CH) / HH,       he = ((i + 1) * CH + HH - 1) / HH;
    int ws = (j * CH) / WW,       we = ((j + 1) * CH + WW - 1) / WW;
    float s = 0.f;
    for (int h = hs; h < he; ++h)
        for (int w = ws; w < we; ++w)
            s += c2[c * CH * CH + h * CH + w];
    comb[(i * WW + j) * 192 + 128 + c] = s / (float)((he - hs) * (we - ws));
}

// ---------------- dt = softplus(zxbcdt[:, 768:772] + dt_bias) ----------------
__global__ void __launch_bounds__(256)
dt_softplus(const float* __restrict__ zx, const float* __restrict__ dtb_in,
            float* __restrict__ dt)
{
    int idx = blockIdx.x * blockDim.x + threadIdx.x;
    if (idx >= LSEQ * NHEADS) return;
    int h = idx & 3, t = idx >> 2;
    dt[idx] = softplus_f(zx[t * D_IN_PROJ + 2 * D_INNER + 2 * D_STATE + h] + dtb_in[h]);
}

// ---------------- depthwise causal conv (k=4) over L on 512 channels + SiLU ----------------
__global__ void __launch_bounds__(256)
dwconv_silu(const float* __restrict__ zx, const float* __restrict__ w,
            const float* __restrict__ b, float* __restrict__ xBC)
{
    int idx = blockIdx.x * blockDim.x + threadIdx.x;
    if (idx >= LSEQ * CONV_DIM) return;
    int c = idx & (CONV_DIM - 1);
    int t = idx >> 9;
    float acc = b[c];
    #pragma unroll
    for (int k = 0; k < D_CONV; ++k) {
        int tt = t + k - (D_CONV - 1);
        if (tt >= 0) acc += zx[tt * D_IN_PROJ + D_INNER + c] * w[c * D_CONV + k];
    }
    xBC[t * CONV_DIM + c] = silu_f(acc);
}

// ---------------- sequential selective scan: one block per head, state in VGPRs ----------------
__global__ void __launch_bounds__(256)
mamba_scan(const float* __restrict__ xBC, const float* __restrict__ dt,
           const float* __restrict__ A_log, const float* __restrict__ Dp,
           float* __restrict__ y)
{
    const int h   = blockIdx.x;        // 0..3
    const int tid = threadIdx.x;
    const int p   = tid >> 2;          // head dim 0..63
    const int nb  = tid & 3;           // state-block 0..3 (32 n's each)
    const float A  = -expf(A_log[h]);
    const float Dh = Dp[h];
    __shared__ alignas(16) float sB[D_STATE];
    __shared__ alignas(16) float sC[D_STATE];
    __shared__ alignas(16) float sx[HEADDIM];
    __shared__ float sdt;

    float st[32];
    #pragma unroll
    for (int i = 0; i < 32; ++i) st[i] = 0.f;

    const int nbase = nb * 32;
    for (int t = 0; t < LSEQ; ++t) {
        const float* __restrict__ rowp = xBC + t * CONV_DIM;
        // vectorized global -> LDS staging (b128 per thread)
        if (tid < 32) {
            *(float4*)(sB + tid * 4) = *(const float4*)(rowp + D_INNER + tid * 4);
        } else if (tid < 64) {
            int q = tid - 32;
            *(float4*)(sC + q * 4) = *(const float4*)(rowp + D_INNER + D_STATE + q * 4);
        } else if (tid < 80) {
            int q = tid - 64;
            *(float4*)(sx + q * 4) = *(const float4*)(rowp + h * HEADDIM + q * 4);
        } else if (tid == 80) {
            sdt = dt[t * NHEADS + h];
        } else if (tid >= 96 && tid < 104 && t + 1 < LSEQ) {
            __builtin_prefetch(rowp + CONV_DIM + (tid - 96) * 64, 0, 1);
        }
        __syncthreads();

        const float dtv  = sdt;
        const float dA   = expf(dtv * A);
        const float coef = dtv * sx[p];
        float ysum = 0.f;
        #pragma unroll
        for (int i4 = 0; i4 < 8; ++i4) {
            const float4 bq = *(const float4*)(sB + nbase + i4 * 4);
            const float4 cq = *(const float4*)(sC + nbase + i4 * 4);
            float s0 = st[i4 * 4 + 0] * dA + coef * bq.x;
            float s1 = st[i4 * 4 + 1] * dA + coef * bq.y;
            float s2 = st[i4 * 4 + 2] * dA + coef * bq.z;
            float s3 = st[i4 * 4 + 3] * dA + coef * bq.w;
            st[i4 * 4 + 0] = s0; st[i4 * 4 + 1] = s1;
            st[i4 * 4 + 2] = s2; st[i4 * 4 + 3] = s3;
            ysum += s0 * cq.x + s1 * cq.y + s2 * cq.z + s3 * cq.w;
        }
        ysum += __shfl_xor(ysum, 1, 32);
        ysum += __shfl_xor(ysum, 2, 32);
        if (nb == 0)
            y[t * D_INNER + h * HEADDIM + p] = ysum + Dh * sx[p];
        __syncthreads();
    }
}

// ---------------- gated RMSNorm (in place on y): y = rmsnorm(y * silu(z)) * nw ----------------
__global__ void __launch_bounds__(256)
gated_rms(const float* __restrict__ zx, const float* __restrict__ nw,
          float* __restrict__ y)
{
    __shared__ float red[256];
    const int t = blockIdx.x, i = threadIdx.x;
    const float z  = zx[t * D_IN_PROJ + i];
    const float yv = y[t * D_INNER + i] * silu_f(z);
    red[i] = yv * yv;
    __syncthreads();
    for (int s = 128; s > 0; s >>= 1) {
        if (i < s) red[i] += red[i + s];
        __syncthreads();
    }
    const float inv = rsqrtf(red[0] / (float)D_INNER + 1e-5f);
    y[t * D_INNER + i] = yv * inv * nw[i];
}

// ---------------- final classifier 64 -> 17 ----------------
__global__ void __launch_bounds__(256)
cls3_gemm(const float* __restrict__ in, const float* __restrict__ w,
          const float* __restrict__ b, float* __restrict__ out)
{
    int idx = blockIdx.x * blockDim.x + threadIdx.x;
    if (idx >= LSEQ * NUM_CLASSES) return;
    int t = idx / NUM_CLASSES, cl = idx - t * NUM_CLASSES;
    float acc = b[cl];
    #pragma unroll 8
    for (int k = 0; k < 64; ++k) acc += in[t * 64 + k] * w[k * NUM_CLASSES + cl];
    out[idx] = acc;
}

// =====================================================================================
extern "C" void kernel_launch(void* const* d_in, const int* in_sizes, int n_in,
                              void* d_out, int out_size, void* d_ws, size_t ws_size,
                              hipStream_t stream)
{
    const float* x        = (const float*)d_in[0];
    const float* ln_pre_g = (const float*)d_in[1];
    const float* ln_pre_b = (const float*)d_in[2];
    const float* fe_w1    = (const float*)d_in[3];
    const float* fe_b1    = (const float*)d_in[4];
    const float* fe_w2    = (const float*)d_in[5];
    const float* fe_b2    = (const float*)d_in[6];
    const float* cv_w1    = (const float*)d_in[7];
    const float* cv_b1    = (const float*)d_in[8];
    const float* cv_w2    = (const float*)d_in[9];
    const float* cv_b2    = (const float*)d_in[10];
    const float* mp_w     = (const float*)d_in[11];
    const float* mp_b     = (const float*)d_in[12];
    const float* mp_ln_g  = (const float*)d_in[13];
    const float* mp_ln_b  = (const float*)d_in[14];
    const float* m_in_w   = (const float*)d_in[15];
    const float* m_conv_w = (const float*)d_in[16];
    const float* m_conv_b = (const float*)d_in[17];
    const float* m_dt_bias= (const float*)d_in[18];
    const float* m_A_log  = (const float*)d_in[19];
    const float* m_D      = (const float*)d_in[20];
    const float* m_norm_w = (const float*)d_in[21];
    const float* m_out_w  = (const float*)d_in[22];
    const float* cls_ln_g = (const float*)d_in[23];
    const float* cls_ln_b = (const float*)d_in[24];
    const float* cls_w1   = (const float*)d_in[25];
    const float* cls_b1   = (const float*)d_in[26];
    const float* cls_w2   = (const float*)d_in[27];
    const float* cls_b2   = (const float*)d_in[28];
    const float* cls_w3   = (const float*)d_in[29];
    const float* cls_b3   = (const float*)d_in[30];
    float* outp = (float*)d_out;

    // -------- workspace layout (floats), 256-aligned, with reuse --------
    float* ws = (float*)d_ws;
    size_t off = 0;
    auto alloc = [&](size_t n) {
        float* pp = ws + off;
        off += (n + 255) & ~(size_t)255;
        return pp;
    };
    float* xn   = alloc((size_t)LSEQ * BANDS);        // LN'd input (HWC)
    float* f1   = alloc((size_t)LSEQ * 256);          // fe hidden / later: y buffer
    float* comb = alloc((size_t)LSEQ * 192);          // [f | conv_feat] / later: mout
    float* c1   = alloc((size_t)32 * CH * CH);
    float* c2   = alloc((size_t)64 * CH * CH);        // later: cls hidden2
    float* zx   = alloc((size_t)LSEQ * D_IN_PROJ);
    float* xbc  = alloc((size_t)LSEQ * CONV_DIM);
    float* dtb  = alloc((size_t)LSEQ * NHEADS);
    float* pbuf = xn;    // mp output (xn dead after conv1)
    float* ybuf = f1;    // scan output (f1 dead after fe2)
    float* mout = comb;  // mamba out-proj (comb dead after mp gemm)
    float* clsa = xn;    // cls hidden1 (pbuf dead after in_proj)
    float* clsb = c2;    // cls hidden2 (c2 dead after pool)

    const int mt = (LSEQ + 15) / 16;                 // 1315 M tiles
    const dim3 gblk(128);
    const unsigned gy4 = (unsigned)((mt + 3) / 4);   // 329

    // 1) LN over bands
    ln_rows<<<LSEQ, 128, 0, stream>>>(x, ln_pre_g, ln_pre_b, xn, BANDS, 0);

    // 2) fe1: xn(21025,200) @ (200,256) + b, GELU -> f1
    gemm_wmma_t<200, 1, true><<<dim3(16, gy4), gblk, 0, stream>>>(
        xn, fe_w1, fe_b1, f1, LSEQ, 256, 256);
    // 3) fe2: f1 @ (256,128) + b, GELU -> comb[:, 0:128] (ldc=192)
    gemm_wmma_t<256, 1, true><<<dim3(8, gy4), gblk, 0, stream>>>(
        f1, fe_w2, fe_b2, comb, LSEQ, 128, 192);
    // 4) conv branch
    conv1_gelu<<<(32 * CH * CH + 255) / 256, 256, 0, stream>>>(xn, cv_w1, cv_b1, c1);
    conv2_gelu<<<(64 * CH * CH + 255) / 256, 256, 0, stream>>>(c1, cv_w2, cv_b2, c2);
    pool_to_comb<<<(64 * HH * WW + 255) / 256, 256, 0, stream>>>(c2, comb);

    // 5) fusion: comb(21025,192) @ (192,128) + b -> pbuf ; then LN+GELU in place
    gemm_wmma_t<192, 0, true><<<dim3(8, gy4), gblk, 0, stream>>>(
        comb, mp_w, mp_b, pbuf, LSEQ, 128, 128);
    ln_rows<<<LSEQ, 128, 0, stream>>>(pbuf, mp_ln_g, mp_ln_b, pbuf, 128, 1);

    // 6) mamba in-proj: p(21025,128) @ (128,772) -> zx
    gemm_wmma_t<128, 0, false><<<dim3(49, gy4), gblk, 0, stream>>>(
        pbuf, m_in_w, nullptr, zx, LSEQ, D_IN_PROJ, D_IN_PROJ);
    // 7) dt + depthwise conv + SiLU
    dt_softplus<<<(LSEQ * NHEADS + 255) / 256, 256, 0, stream>>>(zx, m_dt_bias, dtb);
    dwconv_silu<<<(LSEQ * CONV_DIM + 255) / 256, 256, 0, stream>>>(zx, m_conv_w, m_conv_b, xbc);

    // 8) sequential selective scan (one block per head, state in VGPRs)
    mamba_scan<<<NHEADS, 256, 0, stream>>>(xbc, dtb, m_A_log, m_D, ybuf);

    // 9) gated RMSNorm in place
    gated_rms<<<LSEQ, 256, 0, stream>>>(zx, m_norm_w, ybuf);

    // 10) out-proj: y(21025,256) @ (256,128) -> mout ; cls LN in place
    gemm_wmma_t<256, 0, false><<<dim3(8, gy4), gblk, 0, stream>>>(
        ybuf, m_out_w, nullptr, mout, LSEQ, 128, 128);
    ln_rows<<<LSEQ, 128, 0, stream>>>(mout, cls_ln_g, cls_ln_b, mout, 128, 0);

    // 11) classifier MLP
    gemm_wmma_t<128, 1, true><<<dim3(8, gy4), gblk, 0, stream>>>(
        mout, cls_w1, cls_b1, clsa, LSEQ, 128, 128);
    gemm_wmma_t<128, 1, true><<<dim3(4, gy4), gblk, 0, stream>>>(
        clsa, cls_w2, cls_b2, clsb, LSEQ, 64, 64);
    cls3_gemm<<<(LSEQ * NUM_CLASSES + 255) / 256, 256, 0, stream>>>(clsb, cls_w3, cls_b3, outp);

    (void)in_sizes; (void)n_in; (void)out_size; (void)ws_size;
}
